// MixtureAttentionWeightExpert_48120813584586
// MI455X (gfx1250) — matrix operand order repaired and checked by reading, not compiled, not run on hardware
//
#include <hip/hip_runtime.h>

#define B_  8
#define S_  1024
#define H_  768
#define NH_ 12
#define HD_ 64
#define G_  12

typedef __attribute__((ext_vector_type(16))) __bf16 v16bf;
typedef __attribute__((ext_vector_type(8)))  __bf16 v8bf;
typedef __attribute__((ext_vector_type(8)))  float  v8f;

union FragAB { v16bf v; v8bf h[2]; };

// round-to-nearest-even fp32 -> bf16
__device__ __forceinline__ unsigned short f2bf(float f) {
  unsigned u = __float_as_uint(f);
  u += 0x7FFFu + ((u >> 16) & 1u);
  return (unsigned short)(u >> 16);
}

// ---------------------------------------------------------------------------
// Kernel 1: mean over sequence axis.  (8,1024,768) -> (8,768)
// ---------------------------------------------------------------------------
__global__ void mean_kernel(const float* __restrict__ x, float* __restrict__ meanw) {
  int col = blockIdx.x * 256 + threadIdx.x;        // 0..6143
  int b = col / H_, h = col % H_;
  const float* p = x + (size_t)b * S_ * H_ + h;
  float s = 0.f;
  for (int i = 0; i < S_; ++i) s += p[(size_t)i * H_];
  meanw[col] = s * (1.0f / (float)S_);
}

// ---------------------------------------------------------------------------
// Kernel 2: gate MLP + softmax + exact JAX threefry2x32 categorical sampling.
// ---------------------------------------------------------------------------
__device__ __forceinline__ void tfround(unsigned& x0, unsigned& x1, int r) {
  x0 += x1; x1 = (x1 << r) | (x1 >> (32 - r)); x1 ^= x0;
}
__device__ __forceinline__ void threefry2x32(unsigned k0, unsigned k1,
                                             unsigned& x0, unsigned& x1) {
  unsigned k2 = k0 ^ k1 ^ 0x1BD11BDAu;
  x0 += k0; x1 += k1;
  tfround(x0,x1,13); tfround(x0,x1,15); tfround(x0,x1,26); tfround(x0,x1,6);
  x0 += k1; x1 += k2 + 1u;
  tfround(x0,x1,17); tfround(x0,x1,29); tfround(x0,x1,16); tfround(x0,x1,24);
  x0 += k2; x1 += k0 + 2u;
  tfround(x0,x1,13); tfround(x0,x1,15); tfround(x0,x1,26); tfround(x0,x1,6);
  x0 += k0; x1 += k1 + 3u;
  tfround(x0,x1,17); tfround(x0,x1,29); tfround(x0,x1,16); tfround(x0,x1,24);
  x0 += k1; x1 += k2 + 4u;
  tfround(x0,x1,13); tfround(x0,x1,15); tfround(x0,x1,26); tfround(x0,x1,6);
  x0 += k2; x1 += k0 + 5u;
}

__global__ void gate_kernel(const float* __restrict__ meanw,
                            const float* __restrict__ W1, const float* __restrict__ b1,
                            const float* __restrict__ W2, const float* __restrict__ b2,
                            float* __restrict__ prob_out, int* __restrict__ idxw) {
  int t = threadIdx.x;
  if (t >= B_ * G_) return;
  int b = t / G_, g = t % G_;
  const float* sp = meanw + b * H_ + g * HD_;

  float h1v[G_];
  for (int k = 0; k < G_; ++k) {
    float s = b1[k];
    for (int d = 0; d < HD_; ++d) s += sp[d] * W1[d * G_ + k];
    h1v[k] = 0.5f * s * (1.0f + erff(s * 0.70710678118654752f));   // exact GELU
  }
  float lg[G_]; float mx = -3.0e38f;
  for (int j = 0; j < G_; ++j) {
    float s = b2[j];
    for (int k = 0; k < G_; ++k) s += h1v[k] * W2[k * G_ + j];
    lg[j] = s; mx = fmaxf(mx, s);
  }
  float pr[G_]; float den = 0.f;
  for (int j = 0; j < G_; ++j) { pr[j] = expf(lg[j] - mx); den += pr[j]; }
  float inv = 1.0f / den;

  int best = 0; float bestv = -3.0e38f;
  for (int j = 0; j < G_; ++j) {
    float p = pr[j] * inv;
    prob_out[t * G_ + j] = p;
    unsigned e = (unsigned)(t * G_ + j);
    unsigned x0, x1;
    if (e < 576u) { x0 = e;        x1 = 576u + e; }
    else          { x0 = e - 576u; x1 = e; }
    threefry2x32(0u, 42u, x0, x1);                 // key(42) -> (hi,lo)=(0,42)
    unsigned bits = (e < 576u) ? x0 : x1;
    float u = __uint_as_float((bits >> 9) | 0x3f800000u) - 1.0f;   // [0,1)
    const float tiny = 1.17549435e-38f;
    u = fmaxf(tiny, u * (1.0f - tiny) + tiny);     // uniform(minval=tiny)
    float score = logf(p) - logf(-logf(u));        // log p + gumbel
    if (score > bestv) { bestv = score; best = j; }
  }
  idxw[t] = best;
}

// ---------------------------------------------------------------------------
// Kernel 3: fp32 value_layer copy + bf16 transposed Vt[b][h][d][k] build.
// ---------------------------------------------------------------------------
__global__ void vconvert_kernel(const float* __restrict__ v,
                                float* __restrict__ vcopy,
                                unsigned short* __restrict__ vt) {
  __shared__ unsigned short ldsT[HD_ * 72];        // [d][k] pitch 72
  const int bh = blockIdx.y, kt = blockIdx.x, t = threadIdx.x;
  const size_t tb = ((size_t)bh * S_ + (size_t)kt * 64) * HD_;
#pragma unroll
  for (int p = 0; p < 4; ++p) {
    int rl = p * 16 + (t >> 4);                    // local k row 0..63
    int c4 = t & 15;                               // float4 column
    size_t off = tb + (size_t)rl * HD_ + c4 * 4;
    float4 d4 = *(const float4*)(v + off);
    *(float4*)(vcopy + off) = d4;                  // fp32 passthrough output
    ldsT[(c4 * 4 + 0) * 72 + rl] = f2bf(d4.x);
    ldsT[(c4 * 4 + 1) * 72 + rl] = f2bf(d4.y);
    ldsT[(c4 * 4 + 2) * 72 + rl] = f2bf(d4.z);
    ldsT[(c4 * 4 + 3) * 72 + rl] = f2bf(d4.w);
  }
  __syncthreads();
  int d = t >> 2, q = t & 3;                       // 64 d-rows x 4 chunks of 16
  const uint4* src = (const uint4*)&ldsT[d * 72 + q * 16];
  uint4 a0 = src[0], a1 = src[1];
  unsigned short* dst = vt + ((size_t)bh * HD_ + d) * S_ + (size_t)kt * 64 + q * 16;
  ((uint4*)dst)[0] = a0;
  ((uint4*)dst)[1] = a1;
}

// ---------------------------------------------------------------------------
// Kernel 4: context GEMM via V_WMMA_F32_16X16X32_BF16.
// 2-deep branchless software pipeline, double-buffered LDS A staging,
// manually unrolled x2 so register-set / buffer selection is compile-time
// constant (no dynamic local indexing -> no scratch spills).
// Per half-step: A frag ds_reads + B loads (L2) -> issue HBM loads for
// tile i+2 -> WMMAs (wait only B) -> convert regs of tile i+1 -> barrier.
// ---------------------------------------------------------------------------
#define PIPE_STEP(KB, RBUF, WBUF, RLD, RST, KBN) do {                          \
    FragAB a;                                                                  \
    a.h[0] = *(const v8bf*)&RBUF[mrow * 40 + koffA];                           \
    a.h[1] = *(const v8bf*)&RBUF[mrow * 40 + 16 + koffA];                      \
    FragAB bb[4];                                                              \
    _Pragma("unroll")                                                          \
    for (int nt = 0; nt < 4; ++nt) {                                           \
      const unsigned short* bp = Bv + ((size_t)(nt * 16 + nl)) * S_ + (KB) + koffB; \
      bb[nt].h[0] = *(const v8bf*)bp;                                          \
      bb[nt].h[1] = *(const v8bf*)(bp + 16);                                   \
    }                                                                          \
    __builtin_amdgcn_sched_barrier(0);   /* B issues before staging */         \
    _Pragma("unroll")                                                          \
    for (int p = 0; p < 4; ++p)                                                \
      RLD[p] = *(const float4*)(Ab + (size_t)(p * 32 + srow) * S_ + (KBN) + scol * 4); \
    __builtin_amdgcn_sched_barrier(0);   /* staging issues before WMMA */      \
    _Pragma("unroll")                                                          \
    for (int nt = 0; nt < 4; ++nt)                                             \
      acc[nt] = __builtin_amdgcn_wmma_f32_16x16x32_bf16(                       \
          false, a.v, false, bb[nt].v, (short)0, acc[nt], false, false);       \
    _Pragma("unroll")                                                          \
    for (int p = 0; p < 4; ++p) {                                              \
      union { unsigned short hh[4]; uint2 u; } pk;                             \
      pk.hh[0] = f2bf(RST[p].x); pk.hh[1] = f2bf(RST[p].y);                    \
      pk.hh[2] = f2bf(RST[p].z); pk.hh[3] = f2bf(RST[p].w);                    \
      *(uint2*)&WBUF[(p * 32 + srow) * 40 + scol * 4] = pk.u;                  \
    }                                                                          \
    __syncthreads();                                                           \
  } while (0)

__global__ void __launch_bounds__(256)
ctx_wmma_kernel(const float* __restrict__ attn,
                const unsigned short* __restrict__ vt,
                const int* __restrict__ idxw,
                float* __restrict__ ctx) {
  __shared__ unsigned short As0[128 * 40];         // double buffer, pitch 40
  __shared__ unsigned short As1[128 * 40];
  const int t = threadIdx.x;
  const int wave = t >> 5, lane = t & 31;
  const int bh = blockIdx.y, mt = blockIdx.x;
  const int b = bh / NH_, h = bh % NH_;
  const int head = idxw[bh];                       // gathered source head
  const float* Ab = attn + (((size_t)(b * NH_ + head)) * S_ + (size_t)mt * 128) * S_;
  const unsigned short* Bv = vt + (size_t)bh * HD_ * S_;

  v8f acc[4] = {};                                 // 16 x 64 fp32 accumulators

  const int mrow  = wave * 16 + (lane & 15);       // A: lane -> M row
  const int koffA = (lane < 16) ? 0 : 8;           // A: lane half -> K lo/hi
  const int nl    = lane & 15;                     // B: lane -> N col
  const int koffB = (lane < 16) ? 0 : 8;           // B: lane half -> K lo/hi

  const int srow = t >> 3;                         // staging: row within pass
  const int scol = t & 7;                          // staging: float4 column
  float4 r0[4], r1[4];                             // two staging sets, static names

  // prologue: tile 0 staged synchronously into As0; tile 1 loads -> r1
#pragma unroll
  for (int p = 0; p < 4; ++p)
    r0[p] = *(const float4*)(Ab + (size_t)(p * 32 + srow) * S_ + scol * 4);
#pragma unroll
  for (int p = 0; p < 4; ++p) {
    union { unsigned short hh[4]; uint2 u; } pk;
    pk.hh[0] = f2bf(r0[p].x); pk.hh[1] = f2bf(r0[p].y);
    pk.hh[2] = f2bf(r0[p].z); pk.hh[3] = f2bf(r0[p].w);
    *(uint2*)&As0[(p * 32 + srow) * 40 + scol * 4] = pk.u;
  }
#pragma unroll
  for (int p = 0; p < 4; ++p)
    r1[p] = *(const float4*)(Ab + (size_t)(p * 32 + srow) * S_ + 32 + scol * 4);
  __syncthreads();

  for (int kb = 0; kb < S_; kb += 64) {
    const int kn0 = (kb + 64 < S_) ? kb + 64 : 0;  // tile i+2 (branchless clamp)
    const int kn1 = (kb + 96 < S_) ? kb + 96 : 0;  // tile i+3
    PIPE_STEP(kb,      As0, As1, r0, r1, kn0);     // compute tile i,   stage i+2
    PIPE_STEP(kb + 32, As1, As0, r1, r0, kn1);     // compute tile i+1, stage i+3
  }

  // Epilogue: C/D layout -> context (b, q, h, d); 64B per half-wave per row.
  const int rbase = (lane < 16) ? 0 : 8;
#pragma unroll
  for (int r = 0; r < 8; ++r) {
    int q = mt * 128 + wave * 16 + rbase + r;
    float* row = ctx + (((size_t)b * S_ + q) * NH_ + h) * HD_;
#pragma unroll
    for (int nt = 0; nt < 4; ++nt) row[nt * 16 + nl] = acc[nt][r];
  }
}

// ---------------------------------------------------------------------------
extern "C" void kernel_launch(void* const* d_in, const int* in_sizes, int n_in,
                              void* d_out, int out_size, void* d_ws, size_t ws_size,
                              hipStream_t stream) {
  const float* x    = (const float*)d_in[0];   // (8,1024,768)
  const float* attn = (const float*)d_in[1];   // (8,12,1024,1024)
  const float* val  = (const float*)d_in[2];   // (8,12,1024,64)
  const float* W1   = (const float*)d_in[3];
  const float* b1   = (const float*)d_in[4];
  const float* W2   = (const float*)d_in[5];
  const float* b2   = (const float*)d_in[6];
  float* out = (float*)d_out;

  // workspace: bf16 Vt (12.6MB) | mean (24KB) | idx (384B)
  const size_t VT_ELEMS = (size_t)B_ * NH_ * HD_ * S_;
  unsigned short* Vt = (unsigned short*)d_ws;
  float* meanw = (float*)((char*)d_ws + VT_ELEMS * sizeof(unsigned short));
  int* idxw = (int*)(meanw + B_ * H_);

  // output tuple: prob (1152) | context (6291456) | value_layer copy (6291456)
  float* prob_out  = out;
  float* ctx_out   = out + (size_t)B_ * G_ * G_;
  float* vcopy_out = ctx_out + (size_t)B_ * S_ * NH_ * HD_;

  mean_kernel<<<(B_ * H_) / 256, 256, 0, stream>>>(x, meanw);
  gate_kernel<<<1, 128, 0, stream>>>(meanw, W1, b1, W2, b2, prob_out, idxw);
  vconvert_kernel<<<dim3(S_ / 64, B_ * NH_), 256, 0, stream>>>(val, vcopy_out, Vt);
  ctx_wmma_kernel<<<dim3(S_ / 128, B_ * NH_), 256, 0, stream>>>(attn, Vt, idxw, ctx_out);
}
